// Unified2DAttentionOnFrameAsym_85950885528318
// MI455X (gfx1250) — matrix operand order repaired and checked by reading, not compiled, not run on hardware
//
#include <hip/hip_runtime.h>
#include <hip/hip_bf16.h>

// ---------------------------------------------------------------------------
// Types for CDNA5 WMMA (wave32): v_wmma_f32_16x16x32_bf16
//   A fragment: 16 bf16 / lane (lane = M row; K = half*16 .. +16 contiguous)
//   B fragment: 16 bf16 / lane (lane = N col; K = half*16 .. +16 contiguous)
//   C/D:        8 f32 / lane  (lane%16 = N col; VGPR i -> row i + 8*(lane/16))
// ---------------------------------------------------------------------------
typedef __bf16 bf16_t;
typedef __attribute__((ext_vector_type(16))) __bf16 v16bf;
typedef __attribute__((ext_vector_type(8)))  __bf16 bf16x8;
typedef __attribute__((ext_vector_type(8)))  float  v8f;
typedef __attribute__((ext_vector_type(4)))  unsigned int u32x4;
typedef __attribute__((ext_vector_type(8)))  int i32x8;
typedef __attribute__((ext_vector_type(4)))  int i32x4;

#define T_LEN  1000
#define T_PAD  1024
#define B_SZ   4
#define NCOL1  8000      // T*F columns per batch for the QKV GEMM
#define CT_CH  512       // QKV output channels
#define C_IN_  256

#if __has_builtin(__builtin_amdgcn_tensor_load_to_lds) && \
    __has_builtin(__builtin_amdgcn_s_wait_tensorcnt)
#define HAVE_TDM 1
#else
#define HAVE_TDM 0
#endif

static __device__ __forceinline__ v8f wmma_bf16(v16bf a, v16bf b, v8f c) {
  return __builtin_amdgcn_wmma_f32_16x16x32_bf16(false, a, false, b, (short)0, c,
                                                 false, false);
}

static __device__ __forceinline__ v16bf ld_frag_split(const bf16_t* p0,
                                                      const bf16_t* p1) {
  bf16x8 lo = *(const bf16x8*)p0;
  bf16x8 hi = *(const bf16x8*)p1;
  v16bf r;
#pragma unroll
  for (int i = 0; i < 8; ++i) { r[i] = lo[i]; r[i + 8] = hi[i]; }
  return r;
}

static __device__ __forceinline__ v16bf ld_frag_contig(const bf16_t* p) {
  return *(const v16bf*)p;   // 32-byte aligned at every call site
}

#if HAVE_TDM
// ---------------------------------------------------------------------------
// 1-D Tensor Data Mover copy: nelem contiguous bf16 from global -> LDS.
// D# per CDNA5 ISA: group0 {count=1, lds_addr,
// global_addr[56:0], type=2}; group1 {data_size=2B, tensor_dim0=tile_dim0=
// nelem, tensor_dim1=tile_dim1=1, tensor_dim0_stride=nelem}; groups 2/3 = 0.
// Flat-pointer truncation gives the LDS byte offset (flat LDS addr[31:0]).
// ---------------------------------------------------------------------------
static __device__ __forceinline__ void tdm_load_1d(const bf16_t* gsrc,
                                                   bf16_t* ldst,
                                                   unsigned nelem) {
  unsigned long long ga = (unsigned long long)(uintptr_t)gsrc;
  unsigned la = (unsigned)(uintptr_t)ldst;
  u32x4 g0;
  g0[0] = 1u;                                      // count=1 (valid user D#)
  g0[1] = la;                                      // lds_addr (bytes)
  g0[2] = (unsigned)(ga & 0xFFFFFFFFu);            // global_addr[31:0]
  g0[3] = (unsigned)((ga >> 32) & 0x01FFFFFFu) | 0x80000000u; // [56:32]|type=2
  i32x8 g1;
  g1[0] = 0x10000;                                 // data_size = 1 -> 2 bytes
  g1[1] = (int)((nelem & 0xFFFFu) << 16);          // tensor_dim0[15:0]
  g1[2] = (int)(0x00010000u | (nelem >> 16));      // tensor_dim0[31:16]|dim1=1
  g1[3] = (int)((nelem & 0xFFFFu) << 16);          // tile_dim0
  g1[4] = 1;                                       // tile_dim1 = 1
  g1[5] = (int)nelem;                              // tensor_dim0_stride
  g1[6] = 0;
  g1[7] = 0;
  i32x4 gz = (i32x4)0;
#if __clang_major__ >= 23
  i32x8 gz8 = (i32x8)0;
  __builtin_amdgcn_tensor_load_to_lds(g0, g1, gz, gz, gz8, 0);
#else
  __builtin_amdgcn_tensor_load_to_lds(g0, g1, gz, gz, 0);
#endif
}
#endif  // HAVE_TDM

// ---------------------------------------------------------------------------
// Kernel 0: fp32 -> bf16 conversions + x transpose to [b][t*8+f][c]
// ---------------------------------------------------------------------------
__global__ void k_convert(const float* __restrict__ x,
                          const float* __restrict__ wqkv,
                          const float* __restrict__ wproj,
                          bf16_t* __restrict__ xb,
                          bf16_t* __restrict__ wq,
                          bf16_t* __restrict__ wp) {
  size_t idx = (size_t)blockIdx.x * blockDim.x + threadIdx.x;
  size_t stride = (size_t)gridDim.x * blockDim.x;
  const size_t NX = (size_t)B_SZ * NCOL1 * C_IN_;
  for (size_t i = idx; i < NX; i += stride) {
    int c = (int)(i & 255u);
    size_t bn = i >> 8;
    int nn = (int)(bn % NCOL1);
    int b  = (int)(bn / NCOL1);
    int t = nn >> 3, f = nn & 7;
    xb[i] = (bf16_t)x[(((size_t)b * C_IN_ + c) * 8 + f) * T_LEN + t];
  }
  for (size_t i = idx; i < (size_t)CT_CH * C_IN_; i += stride)
    wq[i] = (bf16_t)wqkv[i];
  for (size_t i = idx; i < (size_t)C_IN_ * C_IN_; i += stride)
    wp[i] = (bf16_t)wproj[i];
}

// ---------------------------------------------------------------------------
// Kernel 1: QKV GEMM (512x256 x 256x16 tile) + bias + PReLU + channel-LN.
// Activation tile (8 KB, contiguous) staged to LDS once per block via TDM;
// 8 waves read B-fragments from LDS, weights stream from L2.
// ---------------------------------------------------------------------------
__global__ void __launch_bounds__(256) k_qkv_ln(
    const bf16_t* __restrict__ wq, const bf16_t* __restrict__ xb,
    const float* __restrict__ bq, const float* __restrict__ a_s,
    const float* __restrict__ g, const float* __restrict__ be,
    bf16_t* __restrict__ qo, bf16_t* __restrict__ ko,
    bf16_t* __restrict__ vT) {
  const int blk  = blockIdx.x;
  const int b    = blk / 500;
  const int col0 = (blk % 500) * 16;
  const int lane = threadIdx.x & 31;
  const int wave = threadIdx.x >> 5;
  const int n = lane & 15, half = lane >> 4;
  const float alpha = a_s[0];

  const bf16_t* xcol = xb + ((size_t)b * NCOL1 + col0) * C_IN_;
  __builtin_prefetch(wq + (size_t)wave * 64 * C_IN_, 0, 3);

  __shared__ __align__(32) bf16_t xtile[16 * C_IN_];
#if HAVE_TDM
  if (wave == 0) {
    tdm_load_1d(xcol, xtile, 16 * C_IN_);
    __builtin_amdgcn_s_wait_tensorcnt(0);
  }
#else
  {
    const uint4* gsrc = (const uint4*)xcol;
    uint4* ld4 = (uint4*)xtile;
    for (int i = threadIdx.x; i < 512; i += 256) ld4[i] = gsrc[i];
  }
#endif
  __syncthreads();

  v8f acc[4] = {};
#pragma unroll
  for (int kstep = 0; kstep < 8; ++kstep) {
    v16bf Bf = ld_frag_contig(xtile + n * C_IN_ + kstep * 32 + half * 16);
#pragma unroll
    for (int mt = 0; mt < 4; ++mt) {
      const int row = wave * 64 + mt * 16 + n;
      const bf16_t* ap = wq + (size_t)row * C_IN_ + kstep * 32;
      v16bf Af = ld_frag_split(ap + half * 16, ap + half * 16 + 8);
      acc[mt] = wmma_bf16(Af, Bf, acc[mt]);
    }
  }

  // bias + PReLU, per-column LN statistics (each lane's 32 values share col n)
  float vals[4][8];
  float s1 = 0.f, s2 = 0.f;
#pragma unroll
  for (int mt = 0; mt < 4; ++mt)
#pragma unroll
    for (int i = 0; i < 8; ++i) {
      int o = wave * 64 + mt * 16 + i + 8 * half;
      float v = acc[mt][i] + bq[o];
      v = v >= 0.f ? v : alpha * v;
      vals[mt][i] = v;
      s1 += v; s2 += v * v;
    }
  s1 += __shfl_xor(s1, 16, 32);
  s2 += __shfl_xor(s2, 16, 32);
  __shared__ float red[8][16][2];
  __shared__ float stats[16][2];
  if (half == 0) { red[wave][n][0] = s1; red[wave][n][1] = s2; }
  __syncthreads();
  if (threadIdx.x < 16) {
    float t1 = 0.f, t2 = 0.f;
    for (int w = 0; w < 8; ++w) { t1 += red[w][threadIdx.x][0]; t2 += red[w][threadIdx.x][1]; }
    float mu = t1 * (1.f / 512.f);
    float var = t2 * (1.f / 512.f) - mu * mu;
    stats[threadIdx.x][0] = mu;
    stats[threadIdx.x][1] = rsqrtf(var + 1e-5f);
  }
  __syncthreads();
  const float mu = stats[n][0], rs = stats[n][1];
  const int col = col0 + n;
  const int t = col >> 3, f = col & 7;
#pragma unroll
  for (int mt = 0; mt < 4; ++mt)
#pragma unroll
    for (int i = 0; i < 8; ++i) {
      int o = wave * 64 + mt * 16 + i + 8 * half;
      float y = (vals[mt][i] - mu) * rs * g[o] + be[o];
      int grp = o & 63, hh = o >> 6;
      size_t bh = (size_t)b * 8 + hh;
      if (grp < 16) {
        int d = grp * 8 + f;
        qo[(bh * T_LEN + t) * 128 + d] = (bf16_t)y;
      } else if (grp < 32) {
        int d = (grp - 16) * 8 + f;
        ko[(bh * T_LEN + t) * 128 + d] = (bf16_t)y;
      } else {
        int dv = (grp - 32) * 8 + f;
        vT[(bh * 256 + dv) * T_PAD + t] = (bf16_t)y;
      }
    }
}

// ---------------------------------------------------------------------------
// Kernel 2: windowed causal flash attention. One wave = 16 queries.
// 4 chunks of 32 keys cover [t0-112, t0+16) ⊇ window [t0-100, t0+15].
// ---------------------------------------------------------------------------
__global__ void __launch_bounds__(128) k_attn(
    const bf16_t* __restrict__ qi, const bf16_t* __restrict__ ki,
    const bf16_t* __restrict__ vT, bf16_t* __restrict__ ctxB) {
  const int blk = blockIdx.x;
  const int bh  = blk >> 4;
  const int tt  = blk & 15;
  const int b = bh >> 3, h = bh & 7;
  const int wave = threadIdx.x >> 5;
  const int lane = threadIdx.x & 31;
  const int n = lane & 15, half = lane >> 4;
  const int t0 = tt * 64 + wave * 16;

  const bf16_t* qp = qi + (size_t)bh * T_LEN * 128;
  const bf16_t* kp = ki + (size_t)bh * T_LEN * 128;
  const bf16_t* vp = vT + (size_t)bh * 256 * T_PAD;

  v16bf Aq[4];
  {
    int qr = t0 + n; if (qr > T_LEN - 1) qr = T_LEN - 1;
#pragma unroll
    for (int dk = 0; dk < 4; ++dk) {
      const bf16_t* ap = qp + (size_t)qr * 128 + dk * 32;
      Aq[dk] = ld_frag_split(ap + half * 16, ap + half * 16 + 8);
    }
  }

  v8f acc[16] = {};
  float mrow[8], lrow[8];
#pragma unroll
  for (int i = 0; i < 8; ++i) { mrow[i] = -1e30f; lrow[i] = 0.f; }

  __shared__ __align__(32) bf16_t pb[4][16 * 32];
  bf16_t* pw = pb[wave];
  const float sc = 0.08838834764831845f;   // 1/sqrt(128)

  for (int ch = 0; ch < 4; ++ch) {
    const int s0 = t0 - 112 + ch * 32;
    v8f S[2] = {};
#pragma unroll
    for (int kt = 0; kt < 2; ++kt) {
      int sr = s0 + kt * 16 + n;
      if (sr < 0) sr = 0;
      if (sr > T_LEN - 1) sr = T_LEN - 1;
      const bf16_t* kbase = kp + (size_t)sr * 128;
#pragma unroll
      for (int dk = 0; dk < 4; ++dk) {
        v16bf Bk = ld_frag_contig(kbase + dk * 32 + half * 16);
        S[kt] = wmma_bf16(Aq[dk], Bk, S[kt]);
      }
    }
    // mask, online softmax (rows of this half live in lanes sharing lane/16)
#pragma unroll
    for (int i = 0; i < 8; ++i) {
      const int qt = t0 + i + 8 * half;
#pragma unroll
      for (int kt = 0; kt < 2; ++kt) {
        const int s = s0 + kt * 16 + n;
        float v = S[kt][i] * sc;
        const bool ok = (s >= 0) && (s <= qt) && (s >= qt - 100) && (qt < T_LEN);
        S[kt][i] = ok ? v : -1e30f;
      }
      float m = fmaxf(S[0][i], S[1][i]);
      m = fmaxf(m, __shfl_xor(m, 8, 32));
      m = fmaxf(m, __shfl_xor(m, 4, 32));
      m = fmaxf(m, __shfl_xor(m, 2, 32));
      m = fmaxf(m, __shfl_xor(m, 1, 32));
      const float nm = fmaxf(mrow[i], m);
      const float scale = __expf(mrow[i] - nm);
      float p0 = __expf(S[0][i] - nm);
      float p1 = __expf(S[1][i] - nm);
      S[0][i] = p0; S[1][i] = p1;
      float ps = p0 + p1;
      ps += __shfl_xor(ps, 8, 32);
      ps += __shfl_xor(ps, 4, 32);
      ps += __shfl_xor(ps, 2, 32);
      ps += __shfl_xor(ps, 1, 32);
      lrow[i] = lrow[i] * scale + ps;
      mrow[i] = nm;
#pragma unroll
      for (int nt = 0; nt < 16; ++nt) acc[nt][i] = acc[nt][i] * scale;
    }
    // transpose P (C-layout) -> A-layout through LDS
#pragma unroll
    for (int i = 0; i < 8; ++i) {
      const int row = i + 8 * half;
      pw[row * 32 + n]      = (bf16_t)S[0][i];
      pw[row * 32 + 16 + n] = (bf16_t)S[1][i];
    }
    __syncthreads();
    v16bf Ap = ld_frag_split(pw + n * 32 + half * 16,
                             pw + n * 32 + half * 16 + 8);
    // ctx += P @ V : B fragment is 16 contiguous keys from transposed V
    int sb = s0 + half * 16;
    if (sb < 0) sb = 0;                    // masked slots carry p == 0
#pragma unroll
    for (int nt = 0; nt < 16; ++nt) {
      const int dv = nt * 16 + n;
      v16bf Bv = ld_frag_contig(vp + (size_t)dv * T_PAD + sb);
      acc[nt] = wmma_bf16(Ap, Bv, acc[nt]);
    }
    __syncthreads();
  }

  float inv[8];
#pragma unroll
  for (int i = 0; i < 8; ++i) inv[i] = 1.f / lrow[i];
#pragma unroll
  for (int nt = 0; nt < 16; ++nt) {
    const int dv = nt * 16 + n;
    const int c = h * 32 + (dv >> 3);
    const int f = dv & 7;
#pragma unroll
    for (int i = 0; i < 8; ++i) {
      const int qt = t0 + i + 8 * half;
      if (qt < T_LEN)
        ctxB[(((size_t)b * 8 + f) * T_LEN + qt) * 256 + c] =
            (bf16_t)(acc[nt][i] * inv[i]);
    }
  }
}

// ---------------------------------------------------------------------------
// Kernel 3: output projection (256x256) + bias + PReLU + channel-LN -> f32 out
// Activation tile staged to LDS via TDM (rows clamped so the DMA stays in
// bounds on the last tile); 4 waves, 256x16 tile per block.
// ---------------------------------------------------------------------------
__global__ void __launch_bounds__(128) k_proj_ln(
    const bf16_t* __restrict__ wp, const bf16_t* __restrict__ ctxB,
    const float* __restrict__ bp, const float* __restrict__ a_s,
    const float* __restrict__ g, const float* __restrict__ be,
    float* __restrict__ out) {
  const int blk = blockIdx.x;
  const int bf  = blk / 63;
  const int t0  = (blk % 63) * 16;
  const int b = bf >> 3, f = bf & 7;
  const int lane = threadIdx.x & 31;
  const int wave = threadIdx.x >> 5;
  const int n = lane & 15, half = lane >> 4;
  const float alpha = a_s[0];

  const bf16_t* xc = ctxB + ((size_t)b * 8 + f) * T_LEN * 256;
  const int t0c = (t0 > T_LEN - 16) ? (T_LEN - 16) : t0;   // keep DMA in bounds
  int tcol = t0 + n; if (tcol > T_LEN - 1) tcol = T_LEN - 1;
  const int ri = tcol - t0c;                               // row within tile
  __builtin_prefetch(wp + (size_t)wave * 64 * 256, 0, 3);

  __shared__ __align__(32) bf16_t xtile[16 * 256];
#if HAVE_TDM
  if (wave == 0) {
    tdm_load_1d(xc + (size_t)t0c * 256, xtile, 16 * 256);
    __builtin_amdgcn_s_wait_tensorcnt(0);
  }
#else
  {
    const uint4* gsrc = (const uint4*)(xc + (size_t)t0c * 256);
    uint4* ld4 = (uint4*)xtile;
    for (int i = threadIdx.x; i < 512; i += 128) ld4[i] = gsrc[i];
  }
#endif
  __syncthreads();

  v8f acc[4] = {};
#pragma unroll
  for (int kstep = 0; kstep < 8; ++kstep) {
    v16bf Bf = ld_frag_contig(xtile + ri * 256 + kstep * 32 + half * 16);
#pragma unroll
    for (int mt = 0; mt < 4; ++mt) {
      const int row = wave * 64 + mt * 16 + n;
      const bf16_t* ap = wp + (size_t)row * 256 + kstep * 32;
      v16bf Af = ld_frag_split(ap + half * 16, ap + half * 16 + 8);
      acc[mt] = wmma_bf16(Af, Bf, acc[mt]);
    }
  }

  float vals[4][8];
  float s1 = 0.f, s2 = 0.f;
#pragma unroll
  for (int mt = 0; mt < 4; ++mt)
#pragma unroll
    for (int i = 0; i < 8; ++i) {
      int o = wave * 64 + mt * 16 + i + 8 * half;
      float v = acc[mt][i] + bp[o];
      v = v >= 0.f ? v : alpha * v;
      vals[mt][i] = v;
      s1 += v; s2 += v * v;
    }
  s1 += __shfl_xor(s1, 16, 32);
  s2 += __shfl_xor(s2, 16, 32);
  __shared__ float red[4][16][2];
  __shared__ float stats[16][2];
  if (half == 0) { red[wave][n][0] = s1; red[wave][n][1] = s2; }
  __syncthreads();
  if (threadIdx.x < 16) {
    float t1 = 0.f, t2 = 0.f;
    for (int w = 0; w < 4; ++w) { t1 += red[w][threadIdx.x][0]; t2 += red[w][threadIdx.x][1]; }
    float mu = t1 * (1.f / 256.f);
    float var = t2 * (1.f / 256.f) - mu * mu;
    stats[threadIdx.x][0] = mu;
    stats[threadIdx.x][1] = rsqrtf(var + 1e-5f);
  }
  __syncthreads();
  const float mu = stats[n][0], rs = stats[n][1];
  const int t = t0 + n;
  if (t < T_LEN) {
#pragma unroll
    for (int mt = 0; mt < 4; ++mt)
#pragma unroll
      for (int i = 0; i < 8; ++i) {
        int o = wave * 64 + mt * 16 + i + 8 * half;
        float y = (vals[mt][i] - mu) * rs * g[o] + be[o];
        out[(((size_t)b * 256 + o) * 8 + f) * T_LEN + t] = y;
      }
  }
}

// ---------------------------------------------------------------------------
// Host-side launcher
// ---------------------------------------------------------------------------
extern "C" void kernel_launch(void* const* d_in, const int* in_sizes, int n_in,
                              void* d_out, int out_size, void* d_ws, size_t ws_size,
                              hipStream_t stream) {
  const float* x       = (const float*)d_in[0];
  const float* w_qkv   = (const float*)d_in[1];
  const float* b_qkv   = (const float*)d_in[2];
  const float* a_qkv   = (const float*)d_in[3];
  const float* g_qkv   = (const float*)d_in[4];
  const float* be_qkv  = (const float*)d_in[5];
  const float* w_proj  = (const float*)d_in[6];
  const float* b_proj  = (const float*)d_in[7];
  const float* a_proj  = (const float*)d_in[8];
  const float* g_proj  = (const float*)d_in[9];
  const float* be_proj = (const float*)d_in[10];
  float* out = (float*)d_out;

  // workspace layout (bytes, 256-aligned)
  char* ws = (char*)d_ws;
  bf16_t* wq   = (bf16_t*)(ws + 0);          //   512*256*2 = 262144
  bf16_t* wp   = (bf16_t*)(ws + 262144);     //   256*256*2 = 131072
  bf16_t* xb   = (bf16_t*)(ws + 393216);     // 4*8000*256*2 = 16384000
  bf16_t* qb   = (bf16_t*)(ws + 16777216);   // 4*8*1000*128*2 = 8192000
  bf16_t* kb   = (bf16_t*)(ws + 24969216);   // 8192000
  bf16_t* vT   = (bf16_t*)(ws + 33161216);   // 4*8*256*1024*2 = 16777216
  bf16_t* ctxB = (bf16_t*)(ws + 49938432);   // 4*8*1000*256*2 = 16384000
  (void)in_sizes; (void)n_in; (void)out_size; (void)ws_size;

  k_convert<<<2048, 256, 0, stream>>>(x, w_qkv, w_proj, xb, wq, wp);
  k_qkv_ln<<<2000, 256, 0, stream>>>(wq, xb, b_qkv, a_qkv, g_qkv, be_qkv,
                                     qb, kb, vT);
  k_attn<<<512, 128, 0, stream>>>(qb, kb, vT, ctxB);
  k_proj_ln<<<2016, 128, 0, stream>>>(wp, ctxB, b_proj, a_proj, g_proj,
                                      be_proj, out);
}
